// NMT_17652315587342
// MI455X (gfx1250) — matrix operand order repaired and checked by reading, not compiled, not run on hardware
//
#include <hip/hip_runtime.h>
#include <hip/hip_bf16.h>
#include <math.h>

// Problem constants
#define BB 64
#define SS 2048
#define HH 1024
#define EE 1024
#define DD 64

typedef __attribute__((ext_vector_type(16))) __bf16 v16bf;
typedef __attribute__((ext_vector_type(8)))  __bf16 v8bf;
typedef __attribute__((ext_vector_type(8)))  float  v8f;

#define WMMA_BF16(a, b, c) \
  __builtin_amdgcn_wmma_f32_16x16x32_bf16(false, (a), false, (b), (short)0, (c), false, false)

// ---- A fragment from a pre-converted bf16 matrix (row-major [M][K]) ----
// ISA layout (16-bit A 16x32): lanes 0-15 = rows; elements 0..7 = K[half*8..+7],
// elements 8..15 = K[16+half*8..+7]. Two 16 B loads per lane, no conversion.
__device__ __forceinline__ v16bf load_a_frag_bf(const __bf16* __restrict__ A,
                                                int m0, int lda, int k0, int lane) {
  const int row = lane & 15, half = lane >> 4;
  const __bf16* base = A + (size_t)(m0 + row) * lda + k0 + half * 8;
  const v8bf lo = *(const v8bf*)base;
  const v8bf hi = *(const v8bf*)(base + 16);
  v16bf a;
#pragma unroll
  for (int i = 0; i < 8; ++i) { a[i] = lo[i]; a[8 + i] = hi[i]; }
  return a;
}

// B fragment from an LDS-staged fp32 tile: 16 rows (N) x 32 floats (K),
// row-major, 128 B per row. ISA layout: lane = column; elements 0..7 hold
// K[half*8..+7], elements 8..15 hold K[16+half*8..+7].
__device__ __forceinline__ v16bf lds_b_frag(const float* __restrict__ buf, int lane) {
  const int col = lane & 15, half = lane >> 4;
  const float* base = buf + col * 32 + half * 8;
  const float4 w0 = ((const float4*)base)[0];
  const float4 w1 = ((const float4*)base)[1];
  const float4 w2 = ((const float4*)(base + 16))[0];
  const float4 w3 = ((const float4*)(base + 16))[1];
  v16bf b;
  b[0]  = (__bf16)w0.x; b[1]  = (__bf16)w0.y; b[2]  = (__bf16)w0.z; b[3]  = (__bf16)w0.w;
  b[4]  = (__bf16)w1.x; b[5]  = (__bf16)w1.y; b[6]  = (__bf16)w1.z; b[7]  = (__bf16)w1.w;
  b[8]  = (__bf16)w2.x; b[9]  = (__bf16)w2.y; b[10] = (__bf16)w2.z; b[11] = (__bf16)w2.w;
  b[12] = (__bf16)w3.x; b[13] = (__bf16)w3.y; b[14] = (__bf16)w3.z; b[15] = (__bf16)w3.w;
  return b;
}

// Stage a 16x32-float weight tile (rows n0..n0+15, cols k0..k0+31 of W[N][K])
// into LDS via the gfx1250 async copy engine (ASYNCcnt-tracked).
// 2 KB tile = 128 chunks of 16 B; each of 32 lanes issues 4 async b128 loads.
__device__ __forceinline__ void async_stage_b(const float* __restrict__ W,
                                              int n0, int K, int k0,
                                              uint32_t lds_base_bytes, int lane) {
#pragma unroll
  for (int i = 0; i < 4; ++i) {
    const int chunk = i * 32 + lane;             // 0..127
    const int r = chunk >> 3;                    // row within tile (0..15)
    const int c = chunk & 7;                     // 16B unit within row
    const float* g = W + (size_t)(n0 + r) * K + k0 + c * 4;
    const uint32_t l = lds_base_bytes + (uint32_t)chunk * 16u;
    asm volatile("global_load_async_to_lds_b128 %0, %1, off"
                 :: "v"(l), "v"((unsigned long long)(uintptr_t)g)
                 : "memory");
  }
}

// Pipelined 64xK x Kx16 panel: async-stage fp32 B tiles (double buffered,
// per-wave), bf16 A from global, WMMA-accumulate into acc0..acc3.
__device__ __forceinline__ void gemm_panel(const __bf16* __restrict__ A,
                                           const float* __restrict__ W,
                                           int K, int n0, int lane,
                                           const float* lb0, const float* lb1,
                                           uint32_t la0, uint32_t la1,
                                           v8f& acc0, v8f& acc1, v8f& acc2, v8f& acc3) {
  async_stage_b(W, n0, K, 0, la0, lane);
  for (int k0 = 0; k0 < K; k0 += 32) {
    const bool even = ((k0 >> 5) & 1) == 0;
    const float* cur = even ? lb0 : lb1;
    if (k0 + 32 < K) {
      async_stage_b(W, n0, K, k0 + 32, even ? la1 : la0, lane);
      asm volatile("s_wait_asynccnt 0x4" ::: "memory");  // current tile landed
    } else {
      asm volatile("s_wait_asynccnt 0x0" ::: "memory");
    }
    v16bf b  = lds_b_frag(cur, lane);
    v16bf a0 = load_a_frag_bf(A,  0, K, k0, lane);
    v16bf a1 = load_a_frag_bf(A, 16, K, k0, lane);
    v16bf a2 = load_a_frag_bf(A, 32, K, k0, lane);
    v16bf a3 = load_a_frag_bf(A, 48, K, k0, lane);
    acc0 = WMMA_BF16(a0, b, acc0);
    acc1 = WMMA_BF16(a1, b, acc1);
    acc2 = WMMA_BF16(a2, b, acc2);
    acc3 = WMMA_BF16(a3, b, acc3);
  }
}

__device__ __forceinline__ float sigmoidf_(float x) { return 1.0f / (1.0f + __expf(-x)); }

// ---- Prologue: fp32 -> bf16 conversion (grid-stride) ----
__global__ void k_cvt(const float* __restrict__ src, __bf16* __restrict__ dst, int n) {
  for (int i = blockIdx.x * 256 + threadIdx.x; i < n; i += gridDim.x * 256)
    dst[i] = (__bf16)src[i];
}

// ---- Kernel 1: gates = x_t @ W_ih^T + h0 @ W_hh^T + b_ih + b_hh  [64 x 4096]
__global__ __launch_bounds__(256) void k_gates(const __bf16* __restrict__ xbf,
                                               const __bf16* __restrict__ hbf,
                                               const float* __restrict__ W_ih,
                                               const float* __restrict__ W_hh,
                                               const float* __restrict__ b_ih,
                                               const float* __restrict__ b_hh,
                                               float* __restrict__ gates) {
  __shared__ __align__(16) float stage[8][2][512];  // per-wave double buffer, 32 KB
  const int lane = threadIdx.x & 31;
  const int wave = threadIdx.x >> 5;
  const int n0   = (blockIdx.x * 8 + wave) * 16;
  const int col  = lane & 15, half = lane >> 4;

  const float* lb0 = stage[wave][0];
  const float* lb1 = stage[wave][1];
  const uint32_t la0 = (uint32_t)(uintptr_t)lb0;
  const uint32_t la1 = (uint32_t)(uintptr_t)lb1;

  v8f acc0 = {}, acc1 = {}, acc2 = {}, acc3 = {};
  gemm_panel(xbf, W_ih, EE + HH, n0, lane, lb0, lb1, la0, la1, acc0, acc1, acc2, acc3);
  gemm_panel(hbf, W_hh, HH,      n0, lane, lb0, lb1, la0, la1, acc0, acc1, acc2, acc3);

  const float bias = b_ih[n0 + col] + b_hh[n0 + col];
  const int n = n0 + col;
#pragma unroll
  for (int r = 0; r < 8; ++r) {
    const int mr = r + 8 * half;
    gates[(size_t)(mr)      * (4 * HH) + n] = acc0[r] + bias;
    gates[(size_t)(mr + 16) * (4 * HH) + n] = acc1[r] + bias;
    gates[(size_t)(mr + 32) * (4 * HH) + n] = acc2[r] + bias;
    gates[(size_t)(mr + 48) * (4 * HH) + n] = acc3[r] + bias;
  }
}

// ---- Kernel 2: LSTM cell elementwise -> yt (fp32 for attention) + ybf (bf16)
__global__ __launch_bounds__(256) void k_lstm(const float* __restrict__ gates,
                                              const float* __restrict__ c0,
                                              float* __restrict__ yt,
                                              __bf16* __restrict__ ybf) {
  const int idx = blockIdx.x * 256 + threadIdx.x;  // B*H = 65536
  if (idx >= BB * HH) return;
  const int b = idx >> 10, h = idx & (HH - 1);
  const float* g = gates + (size_t)b * (4 * HH);
  const float gi = g[h], gf = g[h + HH], gg = g[h + 2 * HH], go = g[h + 3 * HH];
  const float c = sigmoidf_(gf) * c0[idx] + sigmoidf_(gi) * tanhf(gg);
  const float y = sigmoidf_(go) * tanhf(c);
  yt[idx]  = y;
  ybf[idx] = (__bf16)y;
}

// ---- Generic 64xN GEMM: Out = act(A[64xK] @ W[NxK]^T), A pre-converted bf16
template <bool TANH>
__global__ __launch_bounds__(256) void k_gemm64(const __bf16* __restrict__ A,
                                                const float* __restrict__ W,
                                                float* __restrict__ Out,
                                                int K, int N) {
  __shared__ __align__(16) float stage[8][2][512];
  const int lane = threadIdx.x & 31;
  const int wave = threadIdx.x >> 5;
  const int n0   = (blockIdx.x * 8 + wave) * 16;
  const int col  = lane & 15, half = lane >> 4;

  const float* lb0 = stage[wave][0];
  const float* lb1 = stage[wave][1];
  const uint32_t la0 = (uint32_t)(uintptr_t)lb0;
  const uint32_t la1 = (uint32_t)(uintptr_t)lb1;

  v8f acc0 = {}, acc1 = {}, acc2 = {}, acc3 = {};
  gemm_panel(A, W, K, n0, lane, lb0, lb1, la0, la1, acc0, acc1, acc2, acc3);

  const int n = n0 + col;
#pragma unroll
  for (int r = 0; r < 8; ++r) {
    const int mr = r + 8 * half;
    float v0 = acc0[r], v1 = acc1[r], v2 = acc2[r], v3 = acc3[r];
    if (TANH) { v0 = tanhf(v0); v1 = tanhf(v1); v2 = tanhf(v2); v3 = tanhf(v3); }
    Out[(size_t)(mr)      * N + n] = v0;
    Out[(size_t)(mr + 16) * N + n] = v1;
    Out[(size_t)(mr + 32) * N + n] = v2;
    Out[(size_t)(mr + 48) * N + n] = v3;
  }
}

// ---- Kernel 4: p_t = len * sigmoid(tbuf[b] . W_tan2pt); window [left,right)
__global__ __launch_bounds__(256) void k_pos(const float* __restrict__ tbuf,
                                             const float* __restrict__ W_tan2pt,
                                             const int* __restrict__ encode_len,
                                             float* __restrict__ p_out,
                                             int* __restrict__ lr_out) {
  const int b = blockIdx.x, tid = threadIdx.x;
  __shared__ float red[256];
  float acc = 0.f;
  for (int h = tid; h < HH; h += 256) acc += tbuf[(size_t)b * HH + h] * W_tan2pt[h];
  red[tid] = acc;
  __syncthreads();
  for (int s = 128; s > 0; s >>= 1) {
    if (tid < s) red[tid] += red[tid + s];
    __syncthreads();
  }
  if (tid == 0) {
    const int len = encode_len[b];
    const float p = (float)len * sigmoidf_(red[0]);
    const int pf = (int)floorf(p);
    int L = pf - DD; if (L < 0) L = 0;
    int R = pf + DD; if (R > len) R = len;
    p_out[b] = p;
    lr_out[2 * b] = L;
    lr_out[2 * b + 1] = R;
  }
}

// ---- Kernel 5: windowed attention -> ct (bf16, feeds final WMMA GEMM).
// Only [left,right) of encode_h is touched (<=128 rows instead of 2048).
__global__ __launch_bounds__(256) void k_attn(const float* __restrict__ encode_h,
                                              const float* __restrict__ yt,
                                              const float* __restrict__ p_in,
                                              const int* __restrict__ lr,
                                              __bf16* __restrict__ cbf) {
  const int b = blockIdx.x, tid = threadIdx.x;
  const int lane = tid & 31, wave = tid >> 5;
  __shared__ float s_at[2 * DD];
  __shared__ float s_inv;
  const int L = lr[2 * b], R = lr[2 * b + 1];
  const int Wn = R - L;  // <= 128, > 0
  const float p = p_in[b];
  const float* yb = yt + (size_t)b * HH;
  const float* eb = encode_h + ((size_t)b * SS + L) * HH;

  // scores: one wave per window row
  for (int s = wave; s < Wn; s += 8) {
    const float* row = eb + (size_t)s * HH;
    float acc = 0.f;
    for (int h = lane; h < HH; h += 32) acc += yb[h] * row[h];
#pragma unroll
    for (int off = 16; off > 0; off >>= 1) acc += __shfl_xor(acc, off, 32);
    if (lane == 0) s_at[s] = acc;
  }
  __syncthreads();

  // softmax over window (tiny; serial on one lane)
  if (tid == 0) {
    float m = -3.0e38f;
    for (int s = 0; s < Wn; ++s) m = fmaxf(m, s_at[s]);
    float sum = 0.f;
    for (int s = 0; s < Wn; ++s) { float e = __expf(s_at[s] - m); s_at[s] = e; sum += e; }
    s_inv = 1.0f / sum;
  }
  __syncthreads();
  if (tid < Wn) {
    const float d = (float)(L + tid) - p;
    s_at[tid] *= s_inv * __expf(-d * d * (1.0f / (DD * DD / 2.0f)));
  }
  __syncthreads();

  // ct[b, h] = sum_s at[s] * encode_h[b, L+s, h]; thread owns 4 h's
  const int h4 = tid * 4;
  float4 acc = {0.f, 0.f, 0.f, 0.f};
  for (int s = 0; s < Wn; ++s) {
    const float a = s_at[s];
    const float4 v = *(const float4*)(eb + (size_t)s * HH + h4);
    acc.x += a * v.x; acc.y += a * v.y; acc.z += a * v.z; acc.w += a * v.w;
  }
  __bf16* cb = cbf + (size_t)b * HH + h4;
  cb[0] = (__bf16)acc.x; cb[1] = (__bf16)acc.y;
  cb[2] = (__bf16)acc.z; cb[3] = (__bf16)acc.w;
}

extern "C" void kernel_launch(void* const* d_in, const int* in_sizes, int n_in,
                              void* d_out, int out_size, void* d_ws, size_t ws_size,
                              hipStream_t stream) {
  (void)in_sizes; (void)n_in; (void)out_size; (void)ws_size;
  const float* encode_h   = (const float*)d_in[0];
  const float* x_t        = (const float*)d_in[1];
  const float* h0         = (const float*)d_in[2];
  const float* c0         = (const float*)d_in[3];
  const float* W_ih       = (const float*)d_in[4];
  const float* W_hh       = (const float*)d_in[5];
  const float* b_ih       = (const float*)d_in[6];
  const float* b_hh       = (const float*)d_in[7];
  const float* W_ht2tan   = (const float*)d_in[8];
  const float* W_tan2pt   = (const float*)d_in[9];
  const float* W_ct2ht    = (const float*)d_in[10];
  const int*   encode_len = (const int*)d_in[11];

  float*  ws    = (float*)d_ws;
  float*  gates = ws;                          // 64*4096 f32
  float*  yt    = ws + (size_t)BB * 4 * HH;    // 64*1024 f32
  float*  tbuf  = yt + (size_t)BB * HH;        // 64*1024 f32
  float*  pbuf  = tbuf + (size_t)BB * HH;      // 64 f32
  int*    lrbuf = (int*)(pbuf + BB);           // 128 i32
  __bf16* xbf   = (__bf16*)(lrbuf + 2 * BB);   // 64*2048 bf16
  __bf16* hbf   = xbf + (size_t)BB * (EE + HH);// 64*1024 bf16
  __bf16* ybf   = hbf + (size_t)BB * HH;       // 64*1024 bf16
  __bf16* cbf   = ybf + (size_t)BB * HH;       // 64*1024 bf16

  // 0. bf16 copies of the (tiny, highly reused) A-side operands
  k_cvt<<<128, 256, 0, stream>>>(x_t, xbf, BB * (EE + HH));
  k_cvt<<<64, 256, 0, stream>>>(h0, hbf, BB * HH);
  // 1. gates GEMM (WMMA bf16, async-LDS weight staging)
  k_gates<<<(4 * HH) / 128, 256, 0, stream>>>(xbf, hbf, W_ih, W_hh, b_ih, b_hh, gates);
  // 2. LSTM cell -> yt (f32) + ybf (bf16)
  k_lstm<<<(BB * HH) / 256, 256, 0, stream>>>(gates, c0, yt, ybf);
  // 3. tanh(yt @ W_ht2tan^T)
  k_gemm64<true><<<HH / 128, 256, 0, stream>>>(ybf, W_ht2tan, tbuf, HH, HH);
  // 4. predictive position + window
  k_pos<<<BB, 256, 0, stream>>>(tbuf, W_tan2pt, encode_len, pbuf, lrbuf);
  // 5. windowed attention -> cbf (bf16)
  k_attn<<<BB, 256, 0, stream>>>(encode_h, yt, pbuf, lrbuf, cbf);
  // 6. ht = ct @ W_ct2ht^T -> d_out
  k_gemm64<false><<<HH / 128, 256, 0, stream>>>(cbf, W_ct2ht, (float*)d_out, HH, HH);
}
